// MixtureOfExperts_12979391168575
// MI455X (gfx1250) — matrix operand order repaired and checked by reading, not compiled, not run on hardware
//
#include <hip/hip_runtime.h>
#include <hip/hip_bf16.h>
#include <cstdint>
#include <math.h>

// ---- problem constants (match reference) ----
constexpr int kE  = 8;      // experts
constexpr int kD  = 1024;   // model dim
constexpr int kH  = 4096;   // expert hidden dim
constexpr int kMT = 32;     // token tile (2 x WMMA M)
constexpr int kHC = 2048;   // H chunk held in LDS at once

typedef __attribute__((ext_vector_type(16))) __bf16 v16bf;
typedef __attribute__((ext_vector_type(8)))  __bf16 v8bf;
typedef __attribute__((ext_vector_type(4)))  __bf16 v4bf;
typedef __attribute__((ext_vector_type(8)))  float  v8f;
typedef __attribute__((ext_vector_type(4)))  int    v4i;

union FragBF { v16bf v; v8bf h[2]; };

#if defined(__gfx1250__) && __has_builtin(__builtin_amdgcn_global_load_async_to_lds_b128)
#define USE_ASYNC_LDS 1
#define AS3 __attribute__((address_space(3)))
#endif

__device__ __forceinline__ v8f wmma_bf16(const FragBF& a, const FragBF& b, v8f c) {
  return __builtin_amdgcn_wmma_f32_16x16x32_bf16(false, a.v, false, b.v,
                                                 (short)0, c, false, false);
}

// Branch-free tanh-form GELU: 0.5*v*(1+tanh(0.79788456*(v+0.044715*v^3)))
// written as v * sigmoid(2u), with hardware v_rcp_f32 instead of IEEE
// division. Total error ~1e-3 abs -- far below the bf16 rounding applied
// to h before GEMM2, and free of exec-mask divergence.
__device__ __forceinline__ float gelu_f(float v) {
  float u = v * (0.7978845608028654f + 0.035677408136300125f * v * v);
  return v * __builtin_amdgcn_rcpf(1.0f + __expf(-2.0f * u));
}

// ------------------------------------------------------------------
// Router: one wave per token. fp32 logits, top-2, softmax over the 2,
// dense gate[T,E] + per-expert token list append.
// ------------------------------------------------------------------
__global__ void moe_router_kernel(const float* __restrict__ x,
                                  const float* __restrict__ Wr,   // [D,E]
                                  const float* __restrict__ br,   // [E]
                                  const float* __restrict__ gbias,// [E]
                                  float* __restrict__ gate,       // [T,E]
                                  int* __restrict__ counts,       // [E]
                                  int* __restrict__ tok_idx,      // [E,T]
                                  int T) {
  int w = threadIdx.x >> 5;
  int lane = threadIdx.x & 31;
  int t = blockIdx.x * 8 + w;
  if (t >= T) return;

  float acc[kE];
#pragma unroll
  for (int e = 0; e < kE; ++e) acc[e] = 0.f;

  const float* xr = x + (size_t)t * kD;
  for (int k = lane; k < kD; k += 32) {
    float xv = xr[k];
    const float* wr = Wr + (size_t)k * kE;
#pragma unroll
    for (int e = 0; e < kE; ++e) acc[e] += xv * wr[e];
  }
#pragma unroll
  for (int e = 0; e < kE; ++e) {
#pragma unroll
    for (int off = 16; off > 0; off >>= 1)
      acc[e] += __shfl_xor(acc[e], off, 32);
  }

  if (lane == 0) {
    float lg[kE];
#pragma unroll
    for (int e = 0; e < kE; ++e) lg[e] = acc[e] + br[e] + gbias[e];

    int i0 = 0; float v0 = lg[0];
#pragma unroll
    for (int e = 1; e < kE; ++e) { if (lg[e] > v0) { v0 = lg[e]; i0 = e; } }
    int i1 = -1; float v1 = -3.4e38f;
#pragma unroll
    for (int e = 0; e < kE; ++e) { if (e != i0 && lg[e] > v1) { v1 = lg[e]; i1 = e; } }

    float e1 = __expf(v1 - v0);     // softmax over {v0,v1}; v0 is max
    float p0 = 1.f / (1.f + e1);
    float p1 = e1 * p0;

#pragma unroll
    for (int e = 0; e < kE; ++e) {
      float gv = (e == i0) ? p0 : ((e == i1) ? p1 : 0.f);
      gate[(size_t)t * kE + e] = gv;
    }
    int pos0 = atomicAdd(&counts[i0], 1);
    tok_idx[(size_t)i0 * T + pos0] = t;
    int pos1 = atomicAdd(&counts[i1], 1);
    tok_idx[(size_t)i1 * T + pos1] = t;
  }
}

// ------------------------------------------------------------------
// x -> bf16 (row-major copy, so expert kernels can async-DMA rows).
// ------------------------------------------------------------------
__global__ void moe_convert_x(const float* __restrict__ x,
                              __bf16* __restrict__ xb, long n) {
  long i = ((long)blockIdx.x * blockDim.x + threadIdx.x) * 4;
  if (i + 3 >= n) return;
  float4 v = *(const float4*)(x + i);
  v4bf o = {(__bf16)v.x, (__bf16)v.y, (__bf16)v.z, (__bf16)v.w};
  *(v4bf*)(xb + i) = o;
}

// ------------------------------------------------------------------
// Weight swizzle: fp32 [E][K][N] -> bf16 WMMA-B-fragment order
// [E][K/32][N/16][lane(32)][16]; each GEMM lane's B fragment is a
// contiguous 32-byte run. Lane L<16: col=L, k=0..7,16..23; L>=16: col=L-16,
// k=8..15,24..31 (CDNA5 16-bit B layout).
// ------------------------------------------------------------------
__global__ void moe_swizzle_w(const float* __restrict__ W,
                              __bf16* __restrict__ Wb,
                              int K, int N, long total_frags) {
  long f = (long)blockIdx.x * blockDim.x + threadIdx.x;
  if (f >= total_frags) return;
  int KC = K >> 5, NT = N >> 4;
  int L  = (int)(f & 31);
  long f2 = f >> 5;
  int nt = (int)(f2 % NT);
  long f3 = f2 / NT;
  int kc = (int)(f3 % KC);
  int e  = (int)(f3 / KC);

  int col   = nt * 16 + (L & 15);
  int kbase = kc * 32 + ((L < 16) ? 0 : 8);
  const float* src = W + ((size_t)e * K + kbase) * N + col;
  __bf16* dst = Wb + (size_t)f * 16;
#pragma unroll
  for (int j = 0; j < 16; ++j)
    dst[j] = (__bf16)src[(size_t)(j < 8 ? j : j + 8) * N];
}

// ------------------------------------------------------------------
// Fused expert MLP, one workgroup per (expert, 32-token tile).
// H in 2 chunks of 2048: GEMM1 (x@W1+b1 -> gelu) fills a 32x2048 bf16
// LDS buffer; GEMM2 (kc-outer, all 8 N-tiles inner, A loaded once per
// kc) accumulates into 16 persistent v8f accumulators; epilogue does
// gate-scaled f32 atomics. Each B fragment feeds 2 independent WMMA
// chains (token rows 0-15 / 16-31).
// ------------------------------------------------------------------
__global__ void __attribute__((amdgpu_waves_per_eu(2)))
moe_expert_kernel(const __bf16* __restrict__ xb,  // [T][D] bf16
                  const __bf16* __restrict__ W1b, // [E][D/32][H/16][32][16]
                  const float*  __restrict__ b1,  // [E][H]
                  const __bf16* __restrict__ W2b, // [E][H/32][D/16][32][16]
                  const float*  __restrict__ b2,  // [E][D]
                  const float*  __restrict__ gate,// [T][E]
                  const int*    __restrict__ counts,
                  const int*    __restrict__ tok_idx, // [E][T]
                  float*        __restrict__ out,     // [T][D]
                  int T) {
  const int e    = blockIdx.y;
  const int tile = blockIdx.x;
  const int ne   = counts[e];
  if (tile * kMT >= ne) return;

  constexpr int XPITCH = kD + 8;    // 1032 bf16 per xg row
  constexpr int HPITCH = kHC + 8;   // 2056 bf16 per hbuf row

  extern __shared__ char smem[];
  __bf16* xg   = (__bf16*)smem;                                  // 32*XPITCH
  __bf16* hbuf = (__bf16*)(smem + (size_t)kMT * XPITCH * 2);     // 32*HPITCH
  int*    toks = (int*)  (smem + (size_t)kMT * (XPITCH + HPITCH) * 2);
  float*  gts  = (float*)(toks + kMT);

  const int tid = threadIdx.x;

  if (tid < kMT) {
    int idx = tile * kMT + tid;
    int tok = (idx < ne) ? tok_idx[(size_t)e * T + idx] : 0;
    toks[tid] = tok;
    gts[tid]  = (idx < ne) ? gate[(size_t)tok * kE + e] : 0.f;  // pad rows: gate 0
  }
  __syncthreads();

  // ---- gather 32 token rows of xb into LDS (16B chunks, 4096 total) ----
#ifdef USE_ASYNC_LDS
#pragma unroll
  for (int i = 0; i < 16; ++i) {
    int c = tid + i * 256;
    int r = c >> 7, off = (c & 127) * 8;      // 128 x 16B per 2KB row
    const __bf16* g = xb + (size_t)toks[r] * kD + off;
    __bf16* l = xg + r * XPITCH + off;
    __builtin_amdgcn_global_load_async_to_lds_b128((v4i*)g, (AS3 v4i*)l, 0, 0);
  }
  asm volatile("s_wait_asynccnt 0x0" ::: "memory");
#else
  for (int i = tid; i < kMT * (kD / 8); i += 256) {
    int r = i >> 7, off = (i & 127) * 8;
    *(v8bf*)(xg + r * XPITCH + off) = *(const v8bf*)(xb + (size_t)toks[r] * kD + off);
  }
#endif
  __syncthreads();

  const int w     = tid >> 5;
  const int lane  = tid & 31;
  const int half8 = (lane < 16) ? 0 : 8;  // A: K-half select; C: M-half select
  const int lidx  = lane & 15;            // A: row; B/C: col within 16-tile

  v8f accLo[8], accHi[8];
#pragma unroll
  for (int i = 0; i < 8; ++i) {
    accLo[i] = {0.f,0.f,0.f,0.f,0.f,0.f,0.f,0.f};
    accHi[i] = {0.f,0.f,0.f,0.f,0.f,0.f,0.f,0.f};
  }

  const __bf16* w1e = W1b + (size_t)e * (kD / 32) * (kH / 16) * 512;
  const __bf16* w2e = W2b + (size_t)e * (kH / 32) * (kD / 16) * 512;

  const __bf16* apLoX = xg + lidx * XPITCH + half8;
  const __bf16* apHiX = xg + (16 + lidx) * XPITCH + half8;

  for (int hc = 0; hc < kH / kHC; ++hc) {
    // ---- GEMM1 chunk: [32 x D] @ W1[:, chunk] -> gelu -> hbuf ----
    // 2 N-tiles per A-load block: A LDS traffic halved, 4 wmma/A-load.
    for (int ntb = 0; ntb < 8; ++ntb) {
      int nt0 = w + ntb * 16;               // tiles nt0 and nt0+8
      v8f c00 = {0.f,0.f,0.f,0.f,0.f,0.f,0.f,0.f};
      v8f c01 = c00, c10 = c00, c11 = c00;
      const __bf16* bp = w1e + (size_t)(hc * (kHC / 16) + nt0) * 512 + lane * 16;
#pragma unroll 2
      for (int kc = 0; kc < kD / 32; ++kc) {
        FragBF a0, a1, b0, b1f;
        a0.h[0] = *(const v8bf*)(apLoX + kc * 32);
        a0.h[1] = *(const v8bf*)(apLoX + kc * 32 + 16);
        a1.h[0] = *(const v8bf*)(apHiX + kc * 32);
        a1.h[1] = *(const v8bf*)(apHiX + kc * 32 + 16);
        const __bf16* bpk = bp + (size_t)kc * (kH / 16) * 512;
        b0.h[0]  = *(const v8bf*)(bpk);
        b0.h[1]  = *(const v8bf*)(bpk + 8);
        b1f.h[0] = *(const v8bf*)(bpk + 8 * 512);
        b1f.h[1] = *(const v8bf*)(bpk + 8 * 512 + 8);
        c00 = wmma_bf16(a0, b0, c00);
        c01 = wmma_bf16(a1, b0, c01);
        c10 = wmma_bf16(a0, b1f, c10);
        c11 = wmma_bf16(a1, b1f, c11);
      }
#pragma unroll
      for (int j = 0; j < 2; ++j) {
        int nt   = nt0 + j * 8;
        int coll = nt * 16 + lidx;
        float bias = b1[(size_t)e * kH + hc * kHC + coll];
        v8f cl = j ? c10 : c00;
        v8f ch = j ? c11 : c01;
#pragma unroll
        for (int r = 0; r < 8; ++r) {
          hbuf[(half8 + r) * HPITCH + coll]      = (__bf16)gelu_f(cl[r] + bias);
          hbuf[(16 + half8 + r) * HPITCH + coll] = (__bf16)gelu_f(ch[r] + bias);
        }
      }
    }
    __syncthreads();

    // ---- GEMM2 chunk: kc-outer, A loaded once per kc, 16 wmma/iter ----
    {
      const __bf16* apLoH = hbuf + lidx * HPITCH + half8;
      const __bf16* apHiH = hbuf + (16 + lidx) * HPITCH + half8;
      const __bf16* bpbase = w2e
          + ((size_t)(hc * (kHC / 32)) * (kD / 16) + w) * 512 + lane * 16;
      for (int kc = 0; kc < kHC / 32; ++kc) {
        FragBF a0, a1;
        a0.h[0] = *(const v8bf*)(apLoH + kc * 32);
        a0.h[1] = *(const v8bf*)(apLoH + kc * 32 + 16);
        a1.h[0] = *(const v8bf*)(apHiH + kc * 32);
        a1.h[1] = *(const v8bf*)(apHiH + kc * 32 + 16);
        const __bf16* bpk = bpbase + (size_t)kc * (kD / 16) * 512;
#pragma unroll
        for (int i = 0; i < 8; ++i) {        // nt = w + i*8 -> imm offset i*8KB
          FragBF b;
          b.h[0] = *(const v8bf*)(bpk + i * (8 * 512));
          b.h[1] = *(const v8bf*)(bpk + i * (8 * 512) + 8);
          accLo[i] = wmma_bf16(a0, b, accLo[i]);
          accHi[i] = wmma_bf16(a1, b, accHi[i]);
        }
      }
    }
    __syncthreads();    // hbuf reused by next chunk's GEMM1
  }

  // -------- epilogue: bias, gate scale, atomic accumulate --------
#pragma unroll
  for (int i = 0; i < 8; ++i) {
    int nt  = w + i * 8;
    int col = nt * 16 + lidx;
    float bias = b2[(size_t)e * kD + col];
#pragma unroll
    for (int r = 0; r < 8; ++r) {
      int rowLo = half8 + r;
      int rowHi = 16 + half8 + r;
      atomicAdd(&out[(size_t)toks[rowLo] * kD + col], (accLo[i][r] + bias) * gts[rowLo]);
      atomicAdd(&out[(size_t)toks[rowHi] * kD + col], (accHi[i][r] + bias) * gts[rowHi]);
    }
  }
}

// ------------------------------------------------------------------
extern "C" void kernel_launch(void* const* d_in, const int* in_sizes, int n_in,
                              void* d_out, int out_size, void* d_ws, size_t ws_size,
                              hipStream_t stream) {
  const float* x     = (const float*)d_in[0];  // [B,S,D]
  const float* Wr    = (const float*)d_in[1];  // [D,E]
  const float* br    = (const float*)d_in[2];  // [E]
  const float* gbias = (const float*)d_in[3];  // [E]
  const float* W1    = (const float*)d_in[4];  // [E,D,H]
  const float* b1    = (const float*)d_in[5];  // [E,H]
  const float* W2    = (const float*)d_in[6];  // [E,H,D]
  const float* b2    = (const float*)d_in[7];  // [E,D]
  float* out = (float*)d_out;

  const int T = in_sizes[0] / kD;              // tokens (B*S) = 4096

  // workspace partition
  char* ws = (char*)d_ws;
  float* gate   = (float*)ws;  ws += (size_t)T * kE * sizeof(float);
  int* counts   = (int*)ws;    ws += 64 * sizeof(int);
  int* tok_idx  = (int*)ws;    ws += (size_t)kE * T * sizeof(int);
  ws = (char*)(((uintptr_t)ws + 255) & ~(uintptr_t)255);
  __bf16* xb  = (__bf16*)ws;   ws += (size_t)T * kD * sizeof(__bf16);
  __bf16* W1b = (__bf16*)ws;   ws += (size_t)kE * kD * kH * sizeof(__bf16);
  __bf16* W2b = (__bf16*)ws;   ws += (size_t)kE * kH * kD * sizeof(__bf16);

  (void)hipMemsetAsync(counts, 0, 64 * sizeof(int), stream);
  (void)hipMemsetAsync(d_out, 0, (size_t)out_size * sizeof(float), stream);

  moe_router_kernel<<<(T + 7) / 8, 256, 0, stream>>>(
      x, Wr, br, gbias, gate, counts, tok_idx, T);

  long nx = (long)T * kD;
  moe_convert_x<<<(int)((nx / 4 + 255) / 256), 256, 0, stream>>>(x, xb, nx);

  long frags1 = (long)kE * (kD / 32) * (kH / 16) * 32;
  moe_swizzle_w<<<(frags1 + 255) / 256, 256, 0, stream>>>(W1, W1b, kD, kH, frags1);
  long frags2 = (long)kE * (kH / 32) * (kD / 16) * 32;
  moe_swizzle_w<<<(frags2 + 255) / 256, 256, 0, stream>>>(W2, W2b, kH, kD, frags2);

  size_t lds = (size_t)kMT * (kD + 8) * 2     // xg
             + (size_t)kMT * (kHC + 8) * 2    // hbuf
             + kMT * sizeof(int) + kMT * sizeof(float);
  dim3 grid(T / kMT, kE);
  moe_expert_kernel<<<grid, 256, lds, stream>>>(
      xb, W1b, b1, W2b, b2, gate, counts, tok_idx, out, T);
}
